// Parallel_Attention_6485400617373
// MI455X (gfx1250) — matrix operand order repaired and checked by reading, not compiled, not run on hardware
//
#include <hip/hip_runtime.h>

// Parallel attention (live branch only): A = QK^T/sqrt(D), A_f = entmax15(A, axis=-2),
// xf_ = (A_f^T @ Vs) @ Wof^T + bof, xs_ = xf_ + LN(xf).  f16 WMMA / f32 accumulate.
// av_gemm stages its B operand through LDS with the Tensor Data Mover (TENSORcnt),
// double-buffered, with TDM LDS padding to avoid bank conflicts.

typedef __attribute__((ext_vector_type(16))) _Float16 v16h;
typedef __attribute__((ext_vector_type(8)))  _Float16 v8h;
typedef __attribute__((ext_vector_type(8)))  float    v8f;
typedef __attribute__((ext_vector_type(4)))  unsigned int u32x4;
typedef __attribute__((ext_vector_type(8)))  int      i32x8;
typedef __attribute__((ext_vector_type(4)))  int      i32x4;

#define B_SZ 4
#define N_SZ 4096
#define D_SZ 128
#define BN_SZ (B_SZ * N_SZ)

__device__ __forceinline__ v8f wmma16(v16h a, v16h b, v8f c) {
  return __builtin_amdgcn_wmma_f32_16x16x32_f16(false, a, false, b, (short)0, c,
                                                false, false);
}

// A-fragment (16x32 f16, MxK): lane l -> M=l&15, K = c0..c0+15 contiguous, c0=16*(l>>4)
__device__ __forceinline__ v16h load_a_f16(const _Float16* base, int ld, int lane) {
  const int row = lane & 15;
  const int c0  = (lane >> 4) * 16;
  const _Float16* p = base + (size_t)row * ld + c0;
  v8h lo = *(const v8h*)p;
  v8h hi = *(const v8h*)(p + 8);
  v16h r;
#pragma unroll
  for (int j = 0; j < 8; ++j) { r[j] = lo[j]; r[j + 8] = hi[j]; }
  return r;
}

__device__ __forceinline__ v16h load_a_f32(const float* base, int ld, int lane) {
  const int row = lane & 15;
  const int c0  = (lane >> 4) * 16;
  const float* p = base + (size_t)row * ld + c0;
  v16h r;
#pragma unroll
  for (int j = 0; j < 8; ++j) {
    r[j]     = (_Float16)p[j];
    r[j + 8] = (_Float16)p[8 + j];
  }
  return r;
}

// B-fragment (32x16 f16, KxN) from source stored (n, k) k-major:
// lane l: column n = l&15, K = (l>>4)*16 + 0..15 contiguous.
__device__ __forceinline__ v16h load_b_f16(const _Float16* base, int ld, int lane) {
  const int col = lane & 15;
  const int k0  = (lane >> 4) * 16;
  const _Float16* p = base + (size_t)col * ld + k0;
  v8h lo = *(const v8h*)p;
  v8h hi = *(const v8h*)(p + 8);
  v16h r;
#pragma unroll
  for (int j = 0; j < 8; ++j) { r[j] = lo[j]; r[j + 8] = hi[j]; }
  return r;
}

__device__ __forceinline__ v16h load_b_f32(const float* base, int ld, int lane) {
  const int col = lane & 15;
  const int k0  = (lane >> 4) * 16;
  const float* p = base + (size_t)col * ld + k0;
  v16h r;
#pragma unroll
  for (int j = 0; j < 16; ++j) r[j] = (_Float16)p[j];
  return r;
}

// ---- Tensor Data Mover: 2D tile (tile_k x tile_d halves) global -> LDS -------------
// D# per CDNA5 ISA. data_size=2B. LDS pad: +2 DWORDs every
// 16 DWORDs (row stride 32 -> 36 halves) to dodge LDS bank conflicts.
__device__ __forceinline__ void tdm_load_2d_f16(const _Float16* gptr, unsigned lds_off,
                                                unsigned tile_k, unsigned tile_d,
                                                unsigned ten_k, unsigned ten_d,
                                                unsigned stride_k) {
  const unsigned long long ga = (unsigned long long)(uintptr_t)gptr;
  u32x4 g0;
  g0[0] = 1u;                                               // count=1, user descriptor
  g0[1] = lds_off;                                          // lds_addr (bytes)
  g0[2] = (unsigned)(ga & 0xffffffffu);                     // global_addr[31:0]
  g0[3] = (unsigned)((ga >> 32) & 0x01ffffffu) | (2u << 30);// global_addr[56:32]|type=2
  i32x8 g1;
  g1[0] = (int)((1u << 16)          // data_size = 1 -> 2 bytes
              | (1u << 20)          // pad_enable
              | (3u << 22)          // pad_interval: pad after 16 DWORDs (64B)
              | (1u << 25));        // pad_amount: 2 DWORDs (8B = 4 halves)
  g1[1] = (int)((ten_k & 0xffffu) << 16);                         // tensor_dim0[15:0]
  g1[2] = (int)(((ten_k >> 16) & 0xffffu) | ((ten_d & 0xffffu) << 16));
  g1[3] = (int)(((ten_d >> 16) & 0xffffu) | ((tile_k & 0xffffu) << 16)); // tile_dim0
  g1[4] = (int)(tile_d & 0xffffu);                                // tile_dim1, dim2=0
  g1[5] = (int)stride_k;                                          // dim0_stride[31:0]
  g1[6] = 0;
  g1[7] = 0;
  i32x4 z4 = {0, 0, 0, 0};
#if defined(__clang_major__) && __clang_major__ >= 23
  i32x8 z8 = {0, 0, 0, 0, 0, 0, 0, 0};
  __builtin_amdgcn_tensor_load_to_lds(g0, g1, z4, z4, z8, 0);
#else
  __builtin_amdgcn_tensor_load_to_lds(g0, g1, z4, z4, 0);
#endif
}

// ---------------- Kernel 1: fused Q/K/Vs projections (f32 in -> f16 out) ------------
// mode 0: Q = xs@Wq^T+bq ; mode 1: K = xf@Wk^T+bk ; mode 2: Vs^T (transposed store)
__global__ void proj_qkv(const float* __restrict__ xs, const float* __restrict__ xf,
                         const float* __restrict__ Wq, const float* __restrict__ bq,
                         const float* __restrict__ Wk, const float* __restrict__ bk,
                         const float* __restrict__ Wv, const float* __restrict__ bv,
                         _Float16* __restrict__ Qh, _Float16* __restrict__ Kh,
                         _Float16* __restrict__ VsT) {
  const int lane = threadIdx.x & 31;
  const int wv   = threadIdx.x >> 5;
  const int mode = blockIdx.y;
  const int m0   = blockIdx.x * 128 + wv * 16;

  const float* X = (mode == 1) ? xf : xs;
  const float* W = (mode == 0) ? Wq : (mode == 1) ? Wk : Wv;
  const float* bias = (mode == 0) ? bq : (mode == 1) ? bk : bv;

  v8f acc[8];
#pragma unroll
  for (int t = 0; t < 8; ++t)
#pragma unroll
    for (int j = 0; j < 8; ++j) acc[t][j] = 0.f;

#pragma unroll
  for (int kb = 0; kb < D_SZ; kb += 32) {
    v16h a = load_a_f32(X + (size_t)m0 * D_SZ + kb, D_SZ, lane);
#pragma unroll
    for (int nt = 0; nt < 8; ++nt) {
      v16h bf = load_b_f32(W + (size_t)(nt * 16) * D_SZ + kb, D_SZ, lane);
      acc[nt] = wmma16(a, bf, acc[nt]);
    }
  }

  const int col = lane & 15, hiw = lane >> 4;
#pragma unroll
  for (int nt = 0; nt < 8; ++nt) {
    const float bb = bias[nt * 16 + col];
#pragma unroll
    for (int r = 0; r < 8; ++r) {
      const int m = m0 + r + 8 * hiw;
      const _Float16 v = (_Float16)(acc[nt][r] + bb);
      if (mode == 2) {
        // Vs^T layout: (b, d, n)  -> friendly K-contiguous B-fragments in av_gemm
        VsT[((size_t)((m >> 12) * D_SZ + nt * 16 + col)) * N_SZ + (m & (N_SZ - 1))] = v;
      } else {
        _Float16* dst = (mode == 0) ? Qh : Kh;
        dst[(size_t)m * D_SZ + nt * 16 + col] = v;
      }
    }
  }
}

// ---------------- Kernel 2: A = Q K^T / sqrt(D), fp32 output ------------------------
__global__ void qk_gemm(const _Float16* __restrict__ Qh, const _Float16* __restrict__ Kh,
                        float* __restrict__ Aout) {
  const int lane = threadIdx.x & 31;
  const int wv   = threadIdx.x >> 5;
  const int b    = blockIdx.z;
  const int m0   = blockIdx.x * 128 + wv * 16;
  const int n0   = blockIdx.y * 128;
  const _Float16* Qb = Qh + (size_t)b * N_SZ * D_SZ;
  const _Float16* Kb = Kh + (size_t)b * N_SZ * D_SZ;

  v8f acc[8];
#pragma unroll
  for (int t = 0; t < 8; ++t)
#pragma unroll
    for (int j = 0; j < 8; ++j) acc[t][j] = 0.f;

#pragma unroll
  for (int kb = 0; kb < D_SZ; kb += 32) {
    v16h a = load_a_f16(Qb + (size_t)m0 * D_SZ + kb, D_SZ, lane);
#pragma unroll
    for (int nt = 0; nt < 8; ++nt) {
      v16h bf = load_b_f16(Kb + (size_t)(n0 + nt * 16) * D_SZ + kb, D_SZ, lane);
      acc[nt] = wmma16(a, bf, acc[nt]);
    }
  }

  const float scale = 0.08838834764831845f;  // 1/sqrt(128)
  const int col = lane & 15, hiw = lane >> 4;
#pragma unroll
  for (int nt = 0; nt < 8; ++nt)
#pragma unroll
    for (int r = 0; r < 8; ++r) {
      const int m = m0 + r + 8 * hiw;
      Aout[((size_t)b * N_SZ + m) * N_SZ + n0 + nt * 16 + col] = acc[nt][r] * scale;
    }
}

// ---------------- Kernel 3: column-wise entmax15, writes A_f^T in f16 ----------------
// Solves sum_n clip((x-max)/2 - tau, 0)^2 = 1 (defining equation of 1.5-entmax)
// by safeguarded Newton — identical fixed point as the reference's sort algorithm.
__global__ void entmax_col(const float* __restrict__ A, _Float16* __restrict__ AfT) {
  const int tid = threadIdx.x;
  const int col = tid & 31;       // one of 32 columns handled by this block
  const int w   = tid >> 5;       // row-stripe id (8 stripes)
  const int b   = blockIdx.y;
  const int m0  = blockIdx.x * 32;
  const float* Acol = A + (size_t)b * N_SZ * N_SZ + m0 + col;  // element [n][col] at +n*N

  __shared__ float red1[8][32], red2[8][32];
  __shared__ float cmax[32], cS1[32], cS2[32];

  // pass 0: column max
  float mx = -3.0e38f;
  for (int n = w; n < N_SZ; n += 8) mx = fmaxf(mx, Acol[(size_t)n * N_SZ]);
  red1[w][col] = mx;
  __syncthreads();
  if (w == 0) {
    float m2 = red1[0][col];
#pragma unroll
    for (int i = 1; i < 8; ++i) m2 = fmaxf(m2, red1[i][col]);
    cmax[col] = m2;
  }
  __syncthreads();
  mx = cmax[col];

  // Newton + bisection on tau in [-1, 0]
  float lo = -1.f, hi = 0.f, tau = -0.5f;
  for (int it = 0; it < 10; ++it) {
    float s1 = 0.f, s2 = 0.f;
    for (int n = w; n < N_SZ; n += 8) {
      float u = (Acol[(size_t)n * N_SZ] - mx) * 0.5f - tau;
      if (u > 0.f) { s1 += u; s2 += u * u; }
    }
    red1[w][col] = s1; red2[w][col] = s2;
    __syncthreads();
    if (w == 0) {
      float S1 = 0.f, S2 = 0.f;
#pragma unroll
      for (int i = 0; i < 8; ++i) { S1 += red1[i][col]; S2 += red2[i][col]; }
      cS1[col] = S1; cS2[col] = S2;
    }
    __syncthreads();
    const float S1 = cS1[col], S2 = cS2[col];
    if (S2 >= 1.f) lo = tau; else hi = tau;
    float tn = tau + (S2 - 1.f) / (2.f * S1);          // Newton step
    if (!(tn > lo && tn < hi)) tn = 0.5f * (lo + hi);  // bisection safeguard (catches NaN)
    tau = tn;
  }

  // final pass: y = clip((x-max)/2 - tau, 0)^2, transposed through LDS -> A_f^T (m, n)
  __shared__ _Float16 tile[64][40];  // 64 n-rows x 32 m-cols (padded)
  for (int c0 = 0; c0 < N_SZ; c0 += 64) {
#pragma unroll
    for (int k = 0; k < 8; ++k) {
      const int n = c0 + w + 8 * k;
      float u = (Acol[(size_t)n * N_SZ] - mx) * 0.5f - tau;
      u = u > 0.f ? u : 0.f;
      tile[w + 8 * k][col] = (_Float16)(u * u);
    }
    __syncthreads();
    const int ml = tid >> 3, n8 = (tid & 7) * 8;
    v8h o;
#pragma unroll
    for (int j = 0; j < 8; ++j) o[j] = tile[n8 + j][ml];
    *(v8h*)(AfT + ((size_t)(b * N_SZ + m0 + ml)) * N_SZ + c0 + n8) = o;
    __syncthreads();
  }
}

// ---------------- Kernel 4: tmp = A_f^T @ Vs  (K = 4096), TDM-staged B --------------
// Vs^T tile (32 k x 128 d) DMA'd to LDS by the Tensor Data Mover, double-buffered.
// LDS row stride = 36 halves (TDM pad) to spread the 64 LDS banks.
__global__ void av_gemm(const _Float16* __restrict__ AfT, const _Float16* __restrict__ VsT,
                        float* __restrict__ tmp) {
  const int lane = threadIdx.x & 31;
  const int wv   = threadIdx.x >> 5;
  const int b    = blockIdx.y;
  const int m0   = blockIdx.x * 128 + wv * 16;
  const _Float16* Ab = AfT + (size_t)b * N_SZ * N_SZ;
  const _Float16* Vb = VsT + (size_t)b * D_SZ * N_SZ;

  __shared__ _Float16 lsB[2][D_SZ][36];   // [buffer][d][k], 36-half padded rows

  v8f acc[8];
#pragma unroll
  for (int t = 0; t < 8; ++t)
#pragma unroll
    for (int j = 0; j < 8; ++j) acc[t][j] = 0.f;

  // prologue: DMA first B tile
  if (wv == 0) {
    tdm_load_2d_f16(Vb, (unsigned)(uintptr_t)&lsB[0][0][0],
                    32, D_SZ, N_SZ, D_SZ, N_SZ);
    __builtin_amdgcn_s_wait_tensorcnt(0);
  }
  __syncthreads();

  const int col = lane & 15, hiw = lane >> 4;
  const int k0  = hiw * 16;

  for (int kb = 0; kb < N_SZ; kb += 32) {
    const int buf = (kb >> 5) & 1;
    // issue DMA for the next tile into the other buffer
    if (wv == 0 && (kb + 32) < N_SZ) {
      tdm_load_2d_f16(Vb + (kb + 32), (unsigned)(uintptr_t)&lsB[buf ^ 1][0][0],
                      32, D_SZ, N_SZ, D_SZ, N_SZ);
    }

    __builtin_prefetch(Ab + (size_t)m0 * N_SZ + kb + 512, 0, 1);  // global_prefetch
    v16h a = load_a_f16(Ab + (size_t)m0 * N_SZ + kb, N_SZ, lane);
#pragma unroll
    for (int nt = 0; nt < 8; ++nt) {
      const _Float16* p = &lsB[buf][nt * 16 + col][k0];
      v8h lo = *(const v8h*)p;
      v8h hi = *(const v8h*)(p + 8);
      v16h bf;
#pragma unroll
      for (int j = 0; j < 8; ++j) { bf[j] = lo[j]; bf[j + 8] = hi[j]; }
      acc[nt] = wmma16(a, bf, acc[nt]);
    }

    if (wv == 0) __builtin_amdgcn_s_wait_tensorcnt(0);
    __syncthreads();
  }

#pragma unroll
  for (int nt = 0; nt < 8; ++nt)
#pragma unroll
    for (int r = 0; r < 8; ++r) {
      const int m = m0 + r + 8 * hiw;
      tmp[((size_t)b * N_SZ + m) * D_SZ + nt * 16 + col] = acc[nt][r];
    }
}

// ---------------- Kernel 5: xf_ = tmp @ Wof^T + bof (fp32 out) ----------------------
__global__ void proj_out(const float* __restrict__ tmp, const float* __restrict__ Wof,
                         const float* __restrict__ bof, float* __restrict__ out2) {
  const int lane = threadIdx.x & 31;
  const int wv   = threadIdx.x >> 5;
  const int m0   = blockIdx.x * 128 + wv * 16;

  v8f acc[8];
#pragma unroll
  for (int t = 0; t < 8; ++t)
#pragma unroll
    for (int j = 0; j < 8; ++j) acc[t][j] = 0.f;

#pragma unroll
  for (int kb = 0; kb < D_SZ; kb += 32) {
    v16h a = load_a_f32(tmp + (size_t)m0 * D_SZ + kb, D_SZ, lane);
#pragma unroll
    for (int nt = 0; nt < 8; ++nt) {
      v16h bf = load_b_f32(Wof + (size_t)(nt * 16) * D_SZ + kb, D_SZ, lane);
      acc[nt] = wmma16(a, bf, acc[nt]);
    }
  }

  const int col = lane & 15, hiw = lane >> 4;
#pragma unroll
  for (int nt = 0; nt < 8; ++nt) {
    const float bb = bof[nt * 16 + col];
#pragma unroll
    for (int r = 0; r < 8; ++r) {
      const int m = m0 + r + 8 * hiw;
      out2[(size_t)m * D_SZ + nt * 16 + col] = acc[nt][r] + bb;
    }
  }
}

// ---------------- Kernel 6: xs_ = xf_ + LayerNorm(xf) -------------------------------
__global__ void ln_add(const float* __restrict__ xf, const float* __restrict__ xfo,
                       const float* __restrict__ g, const float* __restrict__ be,
                       float* __restrict__ out1) {
  const int lane = threadIdx.x & 31;
  const int wv   = threadIdx.x >> 5;
  const int row  = blockIdx.x * 8 + wv;
  const size_t base = (size_t)row * D_SZ + lane * 4;

  float4 x = *(const float4*)(xf + base);
  float s = x.x + x.y + x.z + x.w;
  float q = x.x * x.x + x.y * x.y + x.z * x.z + x.w * x.w;
#pragma unroll
  for (int off = 16; off >= 1; off >>= 1) {
    s += __shfl_xor(s, off);
    q += __shfl_xor(q, off);
  }
  const float mu  = s * (1.0f / D_SZ);
  const float var = q * (1.0f / D_SZ) - mu * mu;
  const float rs  = rsqrtf(var + 1e-5f);

  float4 y  = *(const float4*)(xfo + base);
  float4 gg = *(const float4*)(g + lane * 4);
  float4 bb = *(const float4*)(be + lane * 4);
  float4 o;
  o.x = y.x + (x.x - mu) * rs * gg.x + bb.x;
  o.y = y.y + (x.y - mu) * rs * gg.y + bb.y;
  o.z = y.z + (x.z - mu) * rs * gg.z + bb.z;
  o.w = y.w + (x.w - mu) * rs * gg.w + bb.w;
  *(float4*)(out1 + base) = o;
}

extern "C" void kernel_launch(void* const* d_in, const int* in_sizes, int n_in,
                              void* d_out, int out_size, void* d_ws, size_t ws_size,
                              hipStream_t stream) {
  const float* xs  = (const float*)d_in[0];
  const float* xf  = (const float*)d_in[1];
  const float* Wq  = (const float*)d_in[2];
  const float* bq  = (const float*)d_in[3];
  const float* Wk  = (const float*)d_in[4];
  const float* bk  = (const float*)d_in[5];
  const float* Wvs = (const float*)d_in[6];
  const float* bvs = (const float*)d_in[7];
  // d_in[8..11] (Wvf,bvf,Wos,bos) and d_in[14..15] (g_s,be_s): dead in reference outputs
  const float* Wof = (const float*)d_in[12];
  const float* bof = (const float*)d_in[13];
  const float* g_f = (const float*)d_in[16];
  const float* be_f = (const float*)d_in[17];

  float* out1 = (float*)d_out;                               // xs_
  float* out2 = out1 + (size_t)BN_SZ * D_SZ;                 // xf_
  float* Aout = out2 + (size_t)BN_SZ * D_SZ;                 // A

  char* w = (char*)d_ws;
  const size_t szProj = (size_t)BN_SZ * D_SZ * sizeof(_Float16);   // 4 MiB
  _Float16* Qh  = (_Float16*)(w);
  _Float16* Kh  = (_Float16*)(w + szProj);
  _Float16* VsT = (_Float16*)(w + 2 * szProj);
  _Float16* AfT = (_Float16*)(w + 3 * szProj);                     // 128 MiB
  float*    tmp = (float*)(w + 3 * szProj + (size_t)BN_SZ * N_SZ * sizeof(_Float16));

  proj_qkv<<<dim3(BN_SZ / 128, 3), 256, 0, stream>>>(xs, xf, Wq, bq, Wk, bk, Wvs, bvs,
                                                     Qh, Kh, VsT);
  qk_gemm<<<dim3(N_SZ / 128, N_SZ / 128, B_SZ), 256, 0, stream>>>(Qh, Kh, Aout);
  entmax_col<<<dim3(N_SZ / 32, B_SZ), 256, 0, stream>>>(Aout, AfT);
  av_gemm<<<dim3(N_SZ / 128, B_SZ), 256, 0, stream>>>(AfT, VsT, tmp);
  proj_out<<<dim3(BN_SZ / 128), 256, 0, stream>>>(tmp, Wof, bof, out2);
  ln_add<<<dim3(BN_SZ / 8), 256, 0, stream>>>(xf, out2, g_f, be_f, out1);
}
